// MaskedAttention_75703093559875
// MI455X (gfx1250) — matrix operand order repaired and checked by reading, not compiled, not run on hardware
//
#include <hip/hip_runtime.h>

typedef __attribute__((ext_vector_type(16))) __bf16 bf16x16;
typedef __attribute__((ext_vector_type(8)))  __bf16 bf16x8;
typedef __attribute__((ext_vector_type(8)))  float  f32x8;

union BFrag { bf16x16 v; bf16x8 h[2]; };

__device__ __forceinline__ bf16x8 load8(const __bf16* p) {
  bf16x8 r; __builtin_memcpy(&r, p, 16); return r;
}

__device__ __forceinline__ f32x8 wmma_bf16(bf16x16 a, bf16x16 b, f32x8 c) {
  return __builtin_amdgcn_wmma_f32_16x16x32_bf16(false, a, false, b, (short)0, c, false, false);
}

__device__ __forceinline__ float rmax16(float x) {
#pragma unroll
  for (int off = 1; off < 16; off <<= 1) x = fmaxf(x, __shfl_xor(x, off, 32));
  return x;
}
__device__ __forceinline__ float rsum16(float x) {
#pragma unroll
  for (int off = 1; off < 16; off <<= 1) x += __shfl_xor(x, off, 32);
  return x;
}

constexpr int Bc = 2, Nc = 2048, Cc = 1024, Hc = 16, Dc = 64;

// ---------------- fp32 -> bf16 conversion ----------------
__global__ void cvt_bf16(const float* __restrict__ in, __bf16* __restrict__ out, int n) {
  int i = blockIdx.x * 256 + threadIdx.x;
  if (i < n) out[i] = (__bf16)in[i];
}

// ---------------- QKV GEMM + pack ----------------
// 32(M) x 64(N) tile per wave, one-k-chunk-ahead register pipeline.
// qkv[m, col] = sum_k x[m,k] * w_qkv[col,k]; scatter q,k->[B,H,N,D], v->[B,H,D,N]
__global__ __launch_bounds__(128) void qkv_gemm_pack(
    const __bf16* __restrict__ xb, const __bf16* __restrict__ wb,
    __bf16* __restrict__ qb, __bf16* __restrict__ kb, __bf16* __restrict__ vTb) {
  int lane = threadIdx.x & 31, wave = threadIdx.x >> 5;
  int wid = blockIdx.x * 4 + wave;
  int mt = wid / 48, nt = wid % 48;          // 128 M-tiles(32) x 48 N-tiles(64)
  int n16 = lane & 15, g = lane >> 4;
  const __bf16* arow0 = xb + (size_t)(mt * 32 + n16) * Cc;
  const __bf16* arow1 = arow0 + (size_t)16 * Cc;
  const __bf16* brows[4];
#pragma unroll
  for (int j = 0; j < 4; ++j)
    brows[j] = wb + (size_t)(nt * 64 + j * 16 + n16) * Cc + g * 8;

  f32x8 acc[2][4] = {};
  BFrag a0c, a1c, bc[4], a0n, a1n, bn[4];
  // prime chunk 0
  a0c.h[0] = load8(arow0 + g * 8); a0c.h[1] = load8(arow0 + g * 8 + 16);
  a1c.h[0] = load8(arow1 + g * 8); a1c.h[1] = load8(arow1 + g * 8 + 16);
#pragma unroll
  for (int j = 0; j < 4; ++j) { bc[j].h[0] = load8(brows[j]); bc[j].h[1] = load8(brows[j] + 16); }

  for (int kk = 0; kk < Cc; kk += 32) {
    int kn = (kk + 32 < Cc) ? kk + 32 : 0;   // branch-free wrap on last iter
    // prefetch next chunk into registers
    a0n.h[0] = load8(arow0 + kn + g * 8); a0n.h[1] = load8(arow0 + kn + g * 8 + 16);
    a1n.h[0] = load8(arow1 + kn + g * 8); a1n.h[1] = load8(arow1 + kn + g * 8 + 16);
#pragma unroll
    for (int j = 0; j < 4; ++j) {
      bn[j].h[0] = load8(brows[j] + kn); bn[j].h[1] = load8(brows[j] + kn + 16);
    }
    if (kk + 64 < Cc) {                      // pull lines 2 chunks ahead
      __builtin_prefetch(arow0 + kk + 64, 0, 3);
#pragma unroll
      for (int j = 0; j < 4; ++j) __builtin_prefetch(brows[j] + kk + 64, 0, 3);
    }
#pragma unroll
    for (int j = 0; j < 4; ++j) {
      acc[0][j] = wmma_bf16(a0c.v, bc[j].v, acc[0][j]);
      acc[1][j] = wmma_bf16(a1c.v, bc[j].v, acc[1][j]);
    }
    a0c = a0n; a1c = a1n;
#pragma unroll
    for (int j = 0; j < 4; ++j) bc[j] = bn[j];
  }
#pragma unroll
  for (int sub = 0; sub < 2; ++sub) {
#pragma unroll
    for (int j = 0; j < 4; ++j) {
      int col = nt * 64 + j * 16 + n16;
      int sec = col >> 10;                   // 0=q 1=k 2=v
      int c = col & 1023;
      int h = c >> 6, d = c & 63;
#pragma unroll
      for (int r = 0; r < 8; ++r) {
        int row = mt * 32 + sub * 16 + r + 8 * g;
        int bb = row >> 11, n = row & 2047;
        size_t bh = (size_t)(bb * Hc + h);
        __bf16 v = (__bf16)acc[sub][j][r];
        if (sec == 0)      qb[(bh * Nc + n) * Dc + d] = v;
        else if (sec == 1) kb[(bh * Nc + n) * Dc + d] = v;
        else               vTb[(bh * Dc + d) * Nc + n] = v;
      }
    }
  }
}

// ---------------- fused masked flash attention ----------------
// wave = 16 query rows x D=64; 32-key blocks, K/V fragments double-buffered in registers
__global__ __launch_bounds__(128) void attn_fused(
    const __bf16* __restrict__ qb, const __bf16* __restrict__ kb,
    const __bf16* __restrict__ vTb, const unsigned char* __restrict__ mask,
    __bf16* __restrict__ ob) {
  __shared__ __bf16 Pt[4 * 16 * 32];
  int lane = threadIdx.x & 31, wave = threadIdx.x >> 5;
  int n16 = lane & 15, g = lane >> 4;
  int bh = blockIdx.x >> 5;                  // b*16+h
  int qblk = blockIdx.x & 31;
  int b = bh >> 4, h = bh & 15;
  int q0 = qblk * 64 + wave * 16;

  const __bf16* qrow = qb + ((size_t)bh * Nc + q0 + n16) * Dc;
  BFrag aq0, aq1;
  aq0.h[0] = load8(qrow + g * 8);       aq0.h[1] = load8(qrow + g * 8 + 16);
  aq1.h[0] = load8(qrow + 32 + g * 8);  aq1.h[1] = load8(qrow + 32 + g * 8 + 16);

  bool mq[8];
#pragma unroll
  for (int r = 0; r < 8; ++r) mq[r] = mask[b * Nc + q0 + r + 8 * g] != 0;

  const float scale = 0.125f;                // D^-0.5
  const float MASKV = -3.402823466e38f;
  float rmax[8], rsum[8];
#pragma unroll
  for (int r = 0; r < 8; ++r) { rmax[r] = MASKV; rsum[r] = 0.f; }
  f32x8 o[4] = {};

  __bf16* P = Pt + wave * (16 * 32);
  const __bf16* kbb = kb + (size_t)bh * Nc * Dc;
  const __bf16* vbb = vTb + (size_t)bh * Dc * Nc;
  const unsigned char* mrow = mask + b * Nc;

  // kf[0..1]: keys 0-15 (d 0-31, 32-63); kf[2..3]: keys 16-31
  BFrag kc[4], vc[4], kn[4], vn[4];
  auto load_kv = [&](int k0, BFrag* kf, BFrag* vf) {
    const __bf16* kr0 = kbb + (size_t)(k0 + n16) * Dc + g * 8;
    kf[0].h[0] = load8(kr0);      kf[0].h[1] = load8(kr0 + 16);
    kf[1].h[0] = load8(kr0 + 32); kf[1].h[1] = load8(kr0 + 48);
    const __bf16* kr1 = kbb + (size_t)(k0 + 16 + n16) * Dc + g * 8;
    kf[2].h[0] = load8(kr1);      kf[2].h[1] = load8(kr1 + 16);
    kf[3].h[0] = load8(kr1 + 32); kf[3].h[1] = load8(kr1 + 48);
#pragma unroll
    for (int t = 0; t < 4; ++t) {
      const __bf16* vr = vbb + (size_t)(t * 16 + n16) * Nc + k0 + g * 8;
      vf[t].h[0] = load8(vr); vf[t].h[1] = load8(vr + 16);
    }
  };

  load_kv(0, kc, vc);
  bool mk0c = mrow[n16] != 0, mk1c = mrow[16 + n16] != 0;

  for (int k0 = 0; k0 < Nc; k0 += 32) {
    int knext = (k0 + 32 < Nc) ? k0 + 32 : 0;
    load_kv(knext, kn, vn);                  // prefetch next block
    bool mk0n = mrow[knext + n16] != 0, mk1n = mrow[knext + 16 + n16] != 0;

    f32x8 s0 = {}, s1 = {};
    s0 = wmma_bf16(aq0.v, kc[0].v, s0);
    s0 = wmma_bf16(aq1.v, kc[1].v, s0);
    s1 = wmma_bf16(aq0.v, kc[2].v, s1);
    s1 = wmma_bf16(aq1.v, kc[3].v, s1);

#pragma unroll
    for (int r = 0; r < 8; ++r) {
      float v0 = (mq[r] && mk0c) ? s0[r] * scale : MASKV;
      float v1 = (mq[r] && mk1c) ? s1[r] * scale : MASKV;
      float nm = fmaxf(rmax[r], rmax16(fmaxf(v0, v1)));
      float alpha = __expf(rmax[r] - nm);
      rmax[r] = nm;
      float p0 = __expf(v0 - nm);
      float p1 = __expf(v1 - nm);
      rsum[r] = rsum[r] * alpha + rsum16(p0 + p1);
#pragma unroll
      for (int t = 0; t < 4; ++t) o[t][r] *= alpha;
      P[(r + 8 * g) * 32 + n16]      = (__bf16)p0;   // C-layout -> LDS row-major
      P[(r + 8 * g) * 32 + 16 + n16] = (__bf16)p1;
    }
    asm volatile("s_wait_dscnt 0" ::: "memory");     // DS in-order per wave
    BFrag ap;                                        // reload as 16x32 A-fragment
    ap.h[0] = load8(P + n16 * 32 + g * 8);
    ap.h[1] = load8(P + n16 * 32 + g * 8 + 16);
#pragma unroll
    for (int t = 0; t < 4; ++t)
      o[t] = wmma_bf16(ap.v, vc[t].v, o[t]);

#pragma unroll
    for (int i = 0; i < 4; ++i) { kc[i] = kn[i]; vc[i] = vn[i]; }
    mk0c = mk0n; mk1c = mk1n;
  }
#pragma unroll
  for (int t = 0; t < 4; ++t) {
#pragma unroll
    for (int r = 0; r < 8; ++r) {
      float val = o[t][r] / rsum[r];
      ob[((size_t)b * Nc + q0 + r + 8 * g) * Cc + h * Dc + t * 16 + n16] = (__bf16)val;
    }
  }
}

// ---------------- output projection + bias ----------------
// 32(M) x 64(N) tile per wave with the same register pipeline
__global__ __launch_bounds__(128) void proj_gemm(
    const __bf16* __restrict__ ab, const __bf16* __restrict__ wb,
    const float* __restrict__ bias, float* __restrict__ out) {
  int lane = threadIdx.x & 31, wave = threadIdx.x >> 5;
  int wid = blockIdx.x * 4 + wave;
  int mt = wid >> 4, nt = wid & 15;          // 128 M-tiles(32) x 16 N-tiles(64)
  int n16 = lane & 15, g = lane >> 4;
  const __bf16* arow0 = ab + (size_t)(mt * 32 + n16) * Cc;
  const __bf16* arow1 = arow0 + (size_t)16 * Cc;
  const __bf16* brows[4];
#pragma unroll
  for (int j = 0; j < 4; ++j)
    brows[j] = wb + (size_t)(nt * 64 + j * 16 + n16) * Cc + g * 8;

  f32x8 acc[2][4] = {};
  BFrag a0c, a1c, bc[4], a0n, a1n, bn[4];
  a0c.h[0] = load8(arow0 + g * 8); a0c.h[1] = load8(arow0 + g * 8 + 16);
  a1c.h[0] = load8(arow1 + g * 8); a1c.h[1] = load8(arow1 + g * 8 + 16);
#pragma unroll
  for (int j = 0; j < 4; ++j) { bc[j].h[0] = load8(brows[j]); bc[j].h[1] = load8(brows[j] + 16); }

  for (int kk = 0; kk < Cc; kk += 32) {
    int kn = (kk + 32 < Cc) ? kk + 32 : 0;
    a0n.h[0] = load8(arow0 + kn + g * 8); a0n.h[1] = load8(arow0 + kn + g * 8 + 16);
    a1n.h[0] = load8(arow1 + kn + g * 8); a1n.h[1] = load8(arow1 + kn + g * 8 + 16);
#pragma unroll
    for (int j = 0; j < 4; ++j) {
      bn[j].h[0] = load8(brows[j] + kn); bn[j].h[1] = load8(brows[j] + kn + 16);
    }
    if (kk + 64 < Cc) {
      __builtin_prefetch(arow0 + kk + 64, 0, 3);
#pragma unroll
      for (int j = 0; j < 4; ++j) __builtin_prefetch(brows[j] + kk + 64, 0, 3);
    }
#pragma unroll
    for (int j = 0; j < 4; ++j) {
      acc[0][j] = wmma_bf16(a0c.v, bc[j].v, acc[0][j]);
      acc[1][j] = wmma_bf16(a1c.v, bc[j].v, acc[1][j]);
    }
    a0c = a0n; a1c = a1n;
#pragma unroll
    for (int j = 0; j < 4; ++j) bc[j] = bn[j];
  }
#pragma unroll
  for (int sub = 0; sub < 2; ++sub) {
#pragma unroll
    for (int j = 0; j < 4; ++j) {
      int col = nt * 64 + j * 16 + n16;
      float bv = bias[col];
#pragma unroll
      for (int r = 0; r < 8; ++r) {
        int row = mt * 32 + sub * 16 + r + 8 * g;
        out[(size_t)row * Cc + col] = acc[sub][j][r] + bv;
      }
    }
  }
}

extern "C" void kernel_launch(void* const* d_in, const int* in_sizes, int n_in,
                              void* d_out, int out_size, void* d_ws, size_t ws_size,
                              hipStream_t stream) {
  const float* x              = (const float*)d_in[0];
  const unsigned char* mask   = (const unsigned char*)d_in[1];  // JAX bool = 1 byte
  const float* w_qkv          = (const float*)d_in[2];
  const float* w_proj         = (const float*)d_in[3];
  const float* b_proj         = (const float*)d_in[4];
  float* out = (float*)d_out;

  char* ws = (char*)d_ws;
  size_t off = 0;
  auto alloc = [&](size_t bytes) {
    void* p = ws + off; off += (bytes + 255) & ~(size_t)255; return p;
  };
  __bf16* xb     = (__bf16*)alloc((size_t)Bc * Nc * Cc * 2);
  __bf16* wqkvb  = (__bf16*)alloc((size_t)3 * Cc * Cc * 2);
  __bf16* wprojb = (__bf16*)alloc((size_t)Cc * Cc * 2);
  __bf16* qb     = (__bf16*)alloc((size_t)Bc * Hc * Nc * Dc * 2);
  __bf16* kb     = (__bf16*)alloc((size_t)Bc * Hc * Nc * Dc * 2);
  __bf16* vTb    = (__bf16*)alloc((size_t)Bc * Hc * Dc * Nc * 2);
  __bf16* ab     = (__bf16*)alloc((size_t)Bc * Nc * Cc * 2);

  cvt_bf16<<<(Bc * Nc * Cc) / 256, 256, 0, stream>>>(x, xb, Bc * Nc * Cc);
  cvt_bf16<<<(3 * Cc * Cc) / 256, 256, 0, stream>>>(w_qkv, wqkvb, 3 * Cc * Cc);
  cvt_bf16<<<(Cc * Cc) / 256, 256, 0, stream>>>(w_proj, wprojb, Cc * Cc);

  qkv_gemm_pack<<<(128 * 48) / 4, 128, 0, stream>>>(xb, wqkvb, qb, kb, vTb);
  attn_fused<<<Bc * Hc * (Nc / 64), 128, 0, stream>>>(qb, kb, vTb, mask, ab);
  proj_gemm<<<(128 * 16) / 4, 128, 0, stream>>>(ab, wprojb, b_proj, out);
}
